// HeteroPULLModel_446676598923
// MI455X (gfx1250) — matrix operand order repaired and checked by reading, not compiled, not run on hardware
//
#include <hip/hip_runtime.h>
#include <hip/hip_bf16.h>
#include <math.h>

// ---------------------------------------------------------------------------
// HGT (heterogeneous graph transformer) forward for MI455X / gfx1250.
// Dense projections use v_wmma_f32_16x16x32_bf16 (wave32 WMMA, f32 accum),
// one wave per 16x128 output strip so the A fragment is loaded once per
// K-step and reused across 8 WMMA issues. Per-head arel/mrel 32x32 transforms
// are folded into the K/V weights; edge softmax is a 3-pass atomic
// segment-softmax that stays resident in the 192MB L2.
// ---------------------------------------------------------------------------

#define NNODES 100000
#define NEDGES 500000
#define NPAIRS 100000
#define HIDW   128
#define OUTW   64
#define NHEADS 4
#define HDIM   32

typedef __attribute__((ext_vector_type(16))) __bf16 v16bf;
typedef __attribute__((ext_vector_type(8)))  float  v8f;

// Branch-free tanh-approx GELU (matches jax.nn.gelu approximate=True):
//   tanh(u) = 1 - 2/(exp(2u)+1), computed with fast v_exp_f32; exact at the
//   saturation limits (exp->inf => tanh=1, exp->0 => tanh=-1), no exec-mask
//   divergence.
__device__ __forceinline__ float gelu_f(float x) {
    const float c = 0.7978845608028654f;       // sqrt(2/pi)
    float u = c * (x + 0.044715f * x * x * x);
    float t = 1.0f - 2.0f / (__expf(2.0f * u) + 1.0f);
    return 0.5f * x * (1.0f + t);
}

__device__ __forceinline__ void atomicMaxFloat(float* addr, float value) {
    if (value >= 0.0f) {
        atomicMax((int*)addr, __float_as_int(value));
    } else {
        atomicMin((unsigned int*)addr, __float_as_uint(value));
    }
}

// ---------------------------------------------------------------------------
// WMMA strip GEMM: Y[M x NT*16] = act(X[M x 128]) @ W[128 x NT*16] + bias
// One wave owns a 16-row x NT*16-col strip: A fragment loaded once per
// K-step (float2 pairs; 16-bit A layout has consecutive-K pairs contiguous),
// reused across NT B-fragments / NT wmma issues. Optional gated-skip
// (g = sigmoid(*gatep)) and ReLU in the epilogue.
// ---------------------------------------------------------------------------
template <int NT, int GELU>
__global__ void wmma_gemm_strip_kernel(const float* __restrict__ X,
                                       const float* __restrict__ W,
                                       const float* __restrict__ bias,
                                       float* __restrict__ Y,
                                       const float* __restrict__ Xskip,
                                       const float* __restrict__ gatep,
                                       int M, int relu_out) {
    constexpr int NC = NT * 16;
    const int wave = threadIdx.x >> 5;
    const int lane = threadIdx.x & 31;
    const int tileM = blockIdx.x * 4 + wave;
    if (tileM * 16 >= M) return;
    const int row = tileM * 16 + (lane & 15);
    const int nlo = lane & 15;            // column within a 16-wide tile
    const int hi  = lane >> 4;            // 0 or 1 (lane half)
    const int kA  = hi ? 8 : 0;           // A-fragment K base within group
    const int kB  = hi ? 16 : 0;          // unused; B fragment also uses kA

    v8f acc[NT];
#pragma unroll
    for (int nt = 0; nt < NT; ++nt) acc[nt] = (v8f){};

    const float* __restrict__ xrow = X + (size_t)row * HIDW;

    for (int k0 = 0; k0 < HIDW; k0 += 32) {
        // A fragment: 16-bit A 16x32 layout (ISA 7.12.2); pairs (2j,2j+1)
        // are contiguous in memory -> float2 loads.
        v16bf a;
#pragma unroll
        for (int j = 0; j < 8; ++j) {
            int kk = k0 + ((j & 4) ? 16 : 0) + kA + ((j & 3) << 1);
            float2 xv = *(const float2*)(xrow + kk);
            if (GELU) { xv.x = gelu_f(xv.x); xv.y = gelu_f(xv.y); }
            a[2 * j]     = (__bf16)xv.x;
            a[2 * j + 1] = (__bf16)xv.y;
        }
        // B fragments per column tile; W row-major [128 x NC].
#pragma unroll
        for (int nt = 0; nt < NT; ++nt) {
            const int col = nt * 16 + nlo;
            v16bf b;
#pragma unroll
            for (int e = 0; e < 16; ++e) {
                int kk = k0 + ((e & 8) ? 16 : 0) + kA + (e & 7);
                b[e] = (__bf16)W[(size_t)kk * NC + col];
            }
            acc[nt] = __builtin_amdgcn_wmma_f32_16x16x32_bf16(
                false, a, false, b, (short)0, acc[nt], false, false);
        }
    }

    // Epilogue: C/D layout — VGPR j -> row (j + 8*hi), col = lane&15
    float g = 0.0f;
    if (Xskip) {
        float s = *gatep;
        g = 1.0f / (1.0f + __expf(-s));
    }
    const int mbase = tileM * 16 + (hi << 3);
#pragma unroll
    for (int nt = 0; nt < NT; ++nt) {
        const int col = nt * 16 + nlo;
        const float bv = bias[col];
#pragma unroll
        for (int j = 0; j < 8; ++j) {
            int m = mbase + j;
            float y = acc[nt][j] + bv;
            if (Xskip) y = g * y + (1.0f - g) * Xskip[(size_t)m * HIDW + col];
            if (relu_out) y = fmaxf(y, 0.0f);
            Y[(size_t)m * NC + col] = y;
        }
    }
}

// ---------------------------------------------------------------------------
// Fold per-head 32x32 relation transform into a 128x128 weight + 128 bias:
//   Wc[k][h*32+c] = sum_d W[k][h*32+d] * R[h][d][c]
//   bc[h*32+c]    = sum_d b[h*32+d]    * R[h][d][c]
// ---------------------------------------------------------------------------
__global__ void combine_w_kernel(const float* __restrict__ W,
                                 const float* __restrict__ bvec,
                                 const float* __restrict__ R,
                                 float* __restrict__ Wc,
                                 float* __restrict__ bc) {
    int idx = blockIdx.x * blockDim.x + threadIdx.x;
    if (idx < HIDW * HIDW) {
        int k = idx >> 7, nn = idx & 127;
        int h = nn >> 5, c = nn & 31;
        float s = 0.0f;
#pragma unroll
        for (int d = 0; d < HDIM; ++d)
            s += W[k * HIDW + h * HDIM + d] * R[(h * HDIM + d) * HDIM + c];
        Wc[k * HIDW + nn] = s;
    } else if (idx < HIDW * (HIDW + 1)) {
        int nn = idx - HIDW * HIDW;
        int h = nn >> 5, c = nn & 31;
        float s = 0.0f;
#pragma unroll
        for (int d = 0; d < HDIM; ++d)
            s += bvec[h * HDIM + d] * R[(h * HDIM + d) * HDIM + c];
        bc[nn] = s;
    }
}

__global__ void fill_kernel(float* __restrict__ p, float v, int n) {
    int i = blockIdx.x * blockDim.x + threadIdx.x;
    int stride = gridDim.x * blockDim.x;
    for (; i < n; i += stride) p[i] = v;
}

// Pass A: a[e,h] = dot(q[dst,h,:], krel[src,h,:]) * prel[h] / sqrt(32);
//         store raw a, atomic-max into amax[dst,h].  Rows are 128B aligned
//         so the 32-float dot uses float4 loads.
__global__ void attn_max_kernel(const int* __restrict__ src,
                                const int* __restrict__ dst,
                                const float* __restrict__ q,
                                const float* __restrict__ kr,
                                const float* __restrict__ prel,
                                float* __restrict__ a_out,
                                float* __restrict__ amax, int E) {
    int t = blockIdx.x * blockDim.x + threadIdx.x;
    int e = t >> 2, h = t & 3;
    if (e >= E) return;
    int s = src[e], d = dst[e];
    const float4* qp = (const float4*)(q  + (size_t)d * HIDW + h * HDIM);
    const float4* kp = (const float4*)(kr + (size_t)s * HIDW + h * HDIM);
    float acc = 0.0f;
#pragma unroll
    for (int c = 0; c < HDIM / 4; ++c) {
        float4 qv = qp[c], kv = kp[c];
        acc += qv.x * kv.x + qv.y * kv.y + qv.z * kv.z + qv.w * kv.w;
    }
    acc *= prel[h] * 0.17677669529663687f;   // 1/sqrt(32)
    a_out[(size_t)e * NHEADS + h] = acc;
    atomicMaxFloat(&amax[(size_t)d * NHEADS + h], acc);
}

// Pass B: ex = exp(a - amax[dst]); store ex in-place; atomic-add den[dst,h].
__global__ void attn_expsum_kernel(const int* __restrict__ dst,
                                   float* __restrict__ a_ex,
                                   const float* __restrict__ amax,
                                   float* __restrict__ den, int E) {
    int t = blockIdx.x * blockDim.x + threadIdx.x;
    int e = t >> 2, h = t & 3;
    if (e >= E) return;
    int d = dst[e];
    float ex = __expf(a_ex[(size_t)e * NHEADS + h] - amax[(size_t)d * NHEADS + h]);
    a_ex[(size_t)e * NHEADS + h] = ex;
    atomicAdd(&den[(size_t)d * NHEADS + h], ex);
}

// Pass C: out[dst, c] += vrel[src, c] * ex[e, c/32] / (den[dst, c/32] + 1e-16)
__global__ void attn_msg_kernel(const int* __restrict__ src,
                                const int* __restrict__ dst,
                                const float* __restrict__ vr,
                                const float* __restrict__ ex,
                                const float* __restrict__ den,
                                float* __restrict__ out, int E) {
    int t = blockIdx.x * blockDim.x + threadIdx.x;
    int e = t >> 7, c = t & 127;
    if (e >= E) return;
    int s = src[e], d = dst[e];
    int h = c >> 5;
    float alpha = ex[(size_t)e * NHEADS + h] /
                  (den[(size_t)d * NHEADS + h] + 1e-16f);
    atomicAdd(&out[(size_t)d * HIDW + c], vr[(size_t)s * HIDW + c] * alpha);
}

// logits[i] = clip(dot64(zC[eli0[i]], zD[eli1[i]]), -10, 10); one wave/pair.
__global__ void score_kernel(const float* __restrict__ zC,
                             const float* __restrict__ zD,
                             const int* __restrict__ eli,
                             float* __restrict__ out, int L) {
    int gtid = blockIdx.x * blockDim.x + threadIdx.x;
    int pair = gtid >> 5;
    int lane = threadIdx.x & 31;
    if (pair >= L) return;
    int ci = eli[pair], di = eli[L + pair];
    const float* a = zC + (size_t)ci * OUTW;
    const float* b = zD + (size_t)di * OUTW;
    float s = a[lane] * b[lane] + a[lane + 32] * b[lane + 32];
#pragma unroll
    for (int off = 16; off; off >>= 1) s += __shfl_down(s, off, 32);
    if (lane == 0) out[pair] = fminf(fmaxf(s, -10.0f), 10.0f);
}

// ---------------------------------------------------------------------------
// Host-side orchestration (all launches on `stream`, graph-capture safe).
// ---------------------------------------------------------------------------
static inline void launch_gemm(const float* X, const float* W, const float* b,
                               float* Y, const float* Xskip, const float* gatep,
                               int M, int NCOLS, int gelu_in, int relu_out,
                               hipStream_t stream) {
    dim3 grid((M / 16 + 3) / 4);
    dim3 blk(128);
    if (NCOLS == 128) {
        if (gelu_in)
            hipLaunchKernelGGL((wmma_gemm_strip_kernel<8, 1>), grid, blk, 0,
                               stream, X, W, b, Y, Xskip, gatep, M, relu_out);
        else
            hipLaunchKernelGGL((wmma_gemm_strip_kernel<8, 0>), grid, blk, 0,
                               stream, X, W, b, Y, Xskip, gatep, M, relu_out);
    } else {  // NCOLS == 64
        hipLaunchKernelGGL((wmma_gemm_strip_kernel<4, 0>), grid, blk, 0,
                           stream, X, W, b, Y, Xskip, gatep, M, relu_out);
    }
}

extern "C" void kernel_launch(void* const* d_in, const int* in_sizes, int n_in,
                              void* d_out, int out_size, void* d_ws, size_t ws_size,
                              hipStream_t stream) {
    (void)in_sizes; (void)n_in; (void)out_size; (void)ws_size;

    const float* emb_C = (const float*)d_in[0];
    const float* emb_D = (const float*)d_in[1];
    const float* kW    = (const float*)d_in[2];
    const float* kb    = (const float*)d_in[3];
    const float* qW    = (const float*)d_in[4];
    const float* qb    = (const float*)d_in[5];
    const float* vW    = (const float*)d_in[6];
    const float* vb    = (const float*)d_in[7];
    const float* aW    = (const float*)d_in[8];
    const float* ab    = (const float*)d_in[9];
    const float* skip  = (const float*)d_in[10];
    const float* arel  = (const float*)d_in[11];
    const float* mrel  = (const float*)d_in[12];
    const float* prel  = (const float*)d_in[13];
    const float* outW  = (const float*)d_in[14];
    const float* outb  = (const float*)d_in[15];
    const int*   ei_cd = (const int*)d_in[16];
    const int*   ei_dc = (const int*)d_in[17];
    const int*   eli   = (const int*)d_in[18];
    float* logits = (float*)d_out;

    // Workspace carve-up
    char* ws = (char*)d_ws;
    size_t off = 0;
    auto alloc = [&](size_t bytes) -> float* {
        float* p = (float*)(ws + off);
        off = (off + bytes + 255) & ~(size_t)255;
        return p;
    };
    const size_t NB = (size_t)NNODES * HIDW * sizeof(float);   // 51.2 MB
    float* big[10];
    for (int i = 0; i < 10; ++i) big[i] = alloc(NB);
    float* exbuf = alloc((size_t)NEDGES * NHEADS * sizeof(float));
    float* amax  = alloc((size_t)NNODES * NHEADS * sizeof(float));
    float* den   = alloc((size_t)NNODES * NHEADS * sizeof(float));
    float* Wc    = alloc((size_t)HIDW * HIDW * sizeof(float));
    float* bc    = alloc((size_t)HIDW * sizeof(float));

    float* q_[2]  = { big[0], big[1] };
    float* kr_[2] = { big[2], big[3] };
    float* vr_[2] = { big[4], big[5] };
    float* o_[2]  = { big[6], big[7] };

    const int combN = HIDW * (HIDW + 1);
    const dim3 combGrid((combN + 255) / 256);
    const dim3 blk256(256);
    const int fillGrid = 2048;

    const float* xcur[2] = { emb_C, emb_D };

    for (int l = 0; l < 2; ++l) {
        // Projections (Q plain, K/V with folded per-head relation transforms)
        for (int t = 0; t < 2; ++t) {
            size_t wo = (size_t)(l * 2 + t) * HIDW * HIDW;
            size_t bo = (size_t)(l * 2 + t) * HIDW;
            size_t ro = (size_t)(l * 2 + t) * NHEADS * HDIM * HDIM;
            launch_gemm(xcur[t], qW + wo, qb + bo, q_[t],
                        nullptr, nullptr, NNODES, HIDW, 0, 0, stream);
            hipLaunchKernelGGL(combine_w_kernel, combGrid, blk256, 0, stream,
                               kW + wo, kb + bo, arel + ro, Wc, bc);
            launch_gemm(xcur[t], Wc, bc, kr_[t],
                        nullptr, nullptr, NNODES, HIDW, 0, 0, stream);
            hipLaunchKernelGGL(combine_w_kernel, combGrid, blk256, 0, stream,
                               vW + wo, vb + bo, mrel + ro, Wc, bc);
            launch_gemm(xcur[t], Wc, bc, vr_[t],
                        nullptr, nullptr, NNODES, HIDW, 0, 0, stream);
        }
        // Attention + message passing per relation (r: src=r, dst=1-r)
        for (int r = 0; r < 2; ++r) {
            int s = r, d = 1 - r;
            const int* ei   = (r == 0) ? ei_cd : ei_dc;
            const int* srcp = ei;
            const int* dstp = ei + NEDGES;
            hipLaunchKernelGGL(fill_kernel, dim3(fillGrid), blk256, 0, stream,
                               amax, -INFINITY, NNODES * NHEADS);
            hipLaunchKernelGGL(fill_kernel, dim3(fillGrid), blk256, 0, stream,
                               den, 0.0f, NNODES * NHEADS);
            hipLaunchKernelGGL(fill_kernel, dim3(fillGrid), blk256, 0, stream,
                               o_[d], 0.0f, NNODES * HIDW);
            int tA = NEDGES * NHEADS;
            hipLaunchKernelGGL(attn_max_kernel, dim3((tA + 255) / 256), blk256, 0,
                               stream, srcp, dstp, q_[d], kr_[s],
                               prel + (size_t)(l * 2 + r) * NHEADS,
                               exbuf, amax, NEDGES);
            hipLaunchKernelGGL(attn_expsum_kernel, dim3((tA + 255) / 256), blk256,
                               0, stream, dstp, exbuf, amax, den, NEDGES);
            long long tC = (long long)NEDGES * HIDW;
            hipLaunchKernelGGL(attn_msg_kernel, dim3((unsigned)((tC + 255) / 256)),
                               blk256, 0, stream, srcp, dstp, vr_[s], exbuf, den,
                               o_[d], NEDGES);
        }
        // Output transform with GELU-in, gated skip, ReLU
        float* nx0 = (l == 0) ? big[8] : big[2];
        float* nx1 = (l == 0) ? big[9] : big[3];
        float* nx[2] = { nx0, nx1 };
        for (int t = 0; t < 2; ++t) {
            size_t wo = (size_t)(l * 2 + t) * HIDW * HIDW;
            size_t bo = (size_t)(l * 2 + t) * HIDW;
            launch_gemm(o_[t], aW + wo, ab + bo, nx[t],
                        xcur[t], skip + (l * 2 + t),
                        NNODES, HIDW, /*gelu_in=*/1, /*relu_out=*/1, stream);
        }
        xcur[0] = nx0; xcur[1] = nx1;
    }

    // Output projections (128 -> 64) and pairwise scoring
    float* zC = big[0];
    float* zD = big[1];
    launch_gemm(xcur[0], outW,               outb,        zC,
                nullptr, nullptr, NNODES, OUTW, 0, 0, stream);
    launch_gemm(xcur[1], outW + HIDW * OUTW, outb + OUTW, zD,
                nullptr, nullptr, NNODES, OUTW, 0, 0, stream);

    long long tS = (long long)NPAIRS * 32;
    hipLaunchKernelGGL(score_kernel, dim3((unsigned)((tS + 255) / 256)), blk256,
                       0, stream, zC, zD, eli, logits, NPAIRS);
}
